// PerDeviceLSTM_24378234372166
// MI455X (gfx1250) — compile-verified, matrix operand order checked by reading, not run on hardware
//
#include <hip/hip_runtime.h>

// ---------------------------------------------------------------------------
// PerDeviceLSTM on MI455X (gfx1250, wave32, WMMA)
//   Pass A: G0[d,t,0:512] = W_ih0[d] @ x_t[d] + b0[d]      (batched WMMA GEMM)
//   Pass B: layer-0 recurrence, W_hh0 fragments register-resident, WMMA GEMV
//   Pass C: G1[d,t,0:512] = W_ih1[d] @ h1_t[d] + b1[d]     (batched WMMA GEMM)
//   Pass D: layer-1 recurrence + fused out[t,d] = h2 . W_out[d] + b_out[d]
// ---------------------------------------------------------------------------

typedef __attribute__((ext_vector_type(16))) _Float16 v16h;
typedef __attribute__((ext_vector_type(8)))  _Float16 v8h;
typedef __attribute__((ext_vector_type(8)))  float    v8f;

#define TSEQ 4096
#define NDEV 32
#define HID  128
#define NG   512   // 4*H

union Frag { v16h v; struct { v8h lo, hi; } p; };

// 16-bit WMMA fragment from row-major f16 LDS tile (ISA 7.12.2 layout):
// lane L holds row (A) / column (B)  = row0 + L%16, K runs:
//   VGPR0-3 : K = kk + (L>=16 ? 8 : 0) + 0..7
//   VGPR4-7 : K = kk + (L>=16 ? 8 : 0) + 16..23
// => two 16-byte ds loads per fragment.
__device__ __forceinline__ v16h ld_frag(const _Float16* base, int stride, int row0, int kk) {
  const int lane = threadIdx.x & 31;
  const _Float16* p = base + (size_t)(row0 + (lane & 15)) * stride + kk + ((lane & 16) >> 1);
  Frag f;
  f.p.lo = *(const v8h*)(p);
  f.p.hi = *(const v8h*)(p + 16);
  return f.v;
}

__device__ __forceinline__ v8f wmma_f16(v16h a, v16h b, v8f c) {
  return __builtin_amdgcn_wmma_f32_16x16x32_f16(false, a, false, b, (short)0, c, false, false);
}

__device__ __forceinline__ float fsig(float x) { return 1.f / (1.f + __expf(-x)); }
__device__ __forceinline__ float ftanh(float x) {
  const float t = __expf(-2.f * fabsf(x));
  return copysignf((1.f - t) / (1.f + t), x);
}

// ---------------------------------------------------------------------------
// Input-projection GEMM: per device d, C[128x128] tile of  A[T x K] * W^T + b
//   PASS_A: A = concat(shared[t,0:64], per_dev[d,t,0:8]), K=72 (padded to 96)
//   else  : A = h1[t,d,0:128] (already f16 in workspace),  K=128
// Grid: (T/128, 512/128, D), 256 threads = 8 waves, each wave: 2x4 subtiles.
// ---------------------------------------------------------------------------
template<int KREAL, int KTILES, bool PASS_A>
__global__ __launch_bounds__(256) void inproj_gemm(
    const float* __restrict__ shared_in,   // [T,64]
    const float* __restrict__ per_dev,     // [D,T,8]
    const _Float16* __restrict__ h1,       // [T,D,128]
    const float* __restrict__ W,           // [D,512,KREAL]
    const float* __restrict__ bias,        // [D,512]
    float* __restrict__ Gout)              // [D,T,512]
{
  constexpr int KP = KTILES * 32;
  constexpr int STRIDE = KP + 8;           // halves; 16B multiple, bank-spread
  extern __shared__ unsigned char smemraw[];
  _Float16* As = (_Float16*)smemraw;               // [128][STRIDE]
  _Float16* Bs = As + 128 * STRIDE;                // [128][STRIDE] (rows = gate col n)

  const int d   = blockIdx.z;
  const int t0  = blockIdx.x * 128;
  const int n0  = blockIdx.y * 128;
  const int tid = threadIdx.x;

  for (int idx = tid; idx < 128 * KP; idx += 256) {
    const int i = idx / KP, k = idx % KP;
    if constexpr (PASS_A) {
      float v;
      if (k < 64)      v = shared_in[(size_t)(t0 + i) * 64 + k];
      else if (k < 72) v = per_dev[((size_t)d * TSEQ + (t0 + i)) * 8 + (k - 64)];
      else             v = 0.f;
      As[i * STRIDE + k] = (_Float16)v;
    } else {
      As[i * STRIDE + k] = h1[((size_t)(t0 + i) * NDEV + d) * HID + k];
    }
  }
  for (int idx = tid; idx < 128 * KP; idx += 256) {
    const int n = idx / KP, k = idx % KP;
    const float v = (k < KREAL) ? W[((size_t)d * NG + (n0 + n)) * KREAL + k] : 0.f;
    Bs[n * STRIDE + k] = (_Float16)v;
  }
  __syncthreads();

  const int w    = tid >> 5;
  const int lane = tid & 31;
  const int mi   = (w & 3) * 32;   // rows (time)
  const int ni   = (w >> 2) * 64;  // cols (gates)

  v8f acc[2][4] = {};
  #pragma unroll
  for (int kt = 0; kt < KTILES; ++kt) {
    const int kk = kt * 32;
    v16h a0 = ld_frag(As, STRIDE, mi,      kk);
    v16h a1 = ld_frag(As, STRIDE, mi + 16, kk);
    v16h b0 = ld_frag(Bs, STRIDE, ni,      kk);
    v16h b1 = ld_frag(Bs, STRIDE, ni + 16, kk);
    v16h b2 = ld_frag(Bs, STRIDE, ni + 32, kk);
    v16h b3 = ld_frag(Bs, STRIDE, ni + 48, kk);
    acc[0][0] = wmma_f16(a0, b0, acc[0][0]);
    acc[0][1] = wmma_f16(a0, b1, acc[0][1]);
    acc[0][2] = wmma_f16(a0, b2, acc[0][2]);
    acc[0][3] = wmma_f16(a0, b3, acc[0][3]);
    acc[1][0] = wmma_f16(a1, b0, acc[1][0]);
    acc[1][1] = wmma_f16(a1, b1, acc[1][1]);
    acc[1][2] = wmma_f16(a1, b2, acc[1][2]);
    acc[1][3] = wmma_f16(a1, b3, acc[1][3]);
  }

  // D layout: VGPR r, lanes 0-15 -> M=r, lanes 16-31 -> M=r+8; N = lane%16
  const int rowadd = (lane >> 4) << 3;
  #pragma unroll
  for (int mm = 0; mm < 2; ++mm) {
    #pragma unroll
    for (int s = 0; s < 4; ++s) {
      const int col  = n0 + ni + s * 16 + (lane & 15);
      const float bv = bias[(size_t)d * NG + col];
      const int trow = t0 + mi + mm * 16 + rowadd;
      float* gp = Gout + ((size_t)d * TSEQ + trow) * NG + col;
      #pragma unroll
      for (int r = 0; r < 8; ++r) gp[(size_t)r * NG] = acc[mm][s][r] + bv;
    }
  }
}

// ---------------------------------------------------------------------------
// Sequential recurrence, one device per workgroup. W_hh is converted to f16
// fragments ONCE and kept register-resident for the whole 4096-step loop
// (16 frags x 8 VGPRs = 128 VGPRs/lane; only 8 waves/WGP so no occupancy
// pressure). Per step the only LDS traffic is the h fragment reload, the
// 512 gate stores and the h writeback. G (input projection) streamed from
// HBM with a 16-step global_prefetch lookahead.
// ---------------------------------------------------------------------------
template<bool FINAL>
__global__ __launch_bounds__(256) void lstm_recur(
    const float* __restrict__ Gin,    // [D,T,512]  precomputed input gates
    const float* __restrict__ Whh,    // [D,512,128]
    _Float16* __restrict__ h1out,     // [T,D,128]  (pass B only)
    const float* __restrict__ Wout,   // [D,128]    (pass D only)
    const float* __restrict__ bout,   // [D]        (pass D only)
    float* __restrict__ out)          // [T,D]      (pass D only)
{
  extern __shared__ unsigned char smemraw[];
  _Float16* Ws    = (_Float16*)smemraw;                               // 512*136 f16 (init only)
  float*    gates = (float*)(smemraw + 512 * 136 * 2);                // 512 f32
  float*    red   = (float*)(smemraw + 512 * 136 * 2 + 2048);         // 128 f32
  _Float16* hbuf  = (_Float16*)(smemraw + 512 * 136 * 2 + 2048 + 512);// 160 f16

  const int d = blockIdx.x;
  const int tid = threadIdx.x, lane = tid & 31, w = tid >> 5;

  const float* Wd = Whh + (size_t)d * NG * HID;
  for (int idx = tid; idx < NG * HID; idx += 256) {
    const int n = idx >> 7, k = idx & 127;
    Ws[n * 136 + k] = (_Float16)Wd[idx];
  }
  if (tid < 160) hbuf[tid] = (_Float16)0.f;
  float c = 0.f, woutk = 0.f;
  if (FINAL && tid < HID) woutk = Wout[(size_t)d * HID + tid];
  __syncthreads();

  const int nbase = w * 64;   // this wave's 64 gate columns

  // Hoist all W_hh fragments for this wave's 4 N-subtiles x 4 K-tiles.
  v16h bfr[4][4];
  #pragma unroll
  for (int s = 0; s < 4; ++s)
    #pragma unroll
    for (int kt = 0; kt < 4; ++kt)
      bfr[s][kt] = ld_frag(Ws, 136, nbase + s * 16, kt * 32);

  const float* Gd = Gin + (size_t)d * TSEQ * NG;

  for (int t = 0; t < TSEQ; ++t) {
    // lookahead prefetch of the 2KB gate row for t+16 (hides HBM latency)
    if (w == 7 && lane < 16 && (t + 16) < TSEQ)
      __builtin_prefetch(Gd + (size_t)(t + 16) * NG + lane * 32, 0, 0);

    // A = 16x32 matrix with only row 0 live (= h), lanes 0/16 carry data
    v16h afr[4];
    #pragma unroll
    for (int kt = 0; kt < 4; ++kt) {
      v16h a = {};
      if ((lane & 15) == 0) {
        const int kb = kt * 32 + ((lane & 16) >> 1);
        Frag f;
        f.p.lo = *(const v8h*)(hbuf + kb);
        f.p.hi = *(const v8h*)(hbuf + kb + 16);
        a = f.v;
      }
      afr[kt] = a;
    }
    #pragma unroll
    for (int s = 0; s < 4; ++s) {
      v8f acc = {};
      #pragma unroll
      for (int kt = 0; kt < 4; ++kt)
        acc = wmma_f16(afr[kt], bfr[s][kt], acc);
      if (lane < 16) gates[nbase + s * 16 + lane] = acc[0];   // row 0 of D tile
    }
    __syncthreads();

    if (tid < HID) {                               // cell update, k = tid
      const float* Gt = Gd + (size_t)t * NG;
      const float ig = gates[tid]           + Gt[tid];
      const float fg = gates[HID + tid]     + Gt[HID + tid];
      const float gg = gates[2 * HID + tid] + Gt[2 * HID + tid];
      const float og = gates[3 * HID + tid] + Gt[3 * HID + tid];
      c = fsig(fg) * c + fsig(ig) * ftanh(gg);
      const float h = fsig(og) * ftanh(c);
      hbuf[tid] = (_Float16)h;
      if constexpr (!FINAL) h1out[((size_t)t * NDEV + d) * HID + tid] = (_Float16)h;
      else                  red[tid] = h * woutk;
    }
    __syncthreads();

    if constexpr (FINAL) {
      if (w == 0) {                                // out[t,d] = h2 . W_out + b
        float s2 = red[lane] + red[lane + 32] + red[lane + 64] + red[lane + 96];
        s2 += __shfl_xor(s2, 16, 32);
        s2 += __shfl_xor(s2, 8, 32);
        s2 += __shfl_xor(s2, 4, 32);
        s2 += __shfl_xor(s2, 2, 32);
        s2 += __shfl_xor(s2, 1, 32);
        if (lane == 0) out[(size_t)t * NDEV + d] = s2 + bout[d];
      }
    }
  }
}

// ---------------------------------------------------------------------------
extern "C" void kernel_launch(void* const* d_in, const int* in_sizes, int n_in,
                              void* d_out, int out_size, void* d_ws, size_t ws_size,
                              hipStream_t stream) {
  (void)in_sizes; (void)n_in; (void)out_size; (void)ws_size;
  const float* shared_in = (const float*)d_in[0]; // [4096,64]
  const float* per_dev   = (const float*)d_in[1]; // [32,4096,8]
  const float* W_ih0     = (const float*)d_in[2]; // [32,512,72]
  const float* W_hh0     = (const float*)d_in[3]; // [32,512,128]
  const float* b0        = (const float*)d_in[4]; // [32,512]
  const float* W_ih1     = (const float*)d_in[5]; // [32,512,128]
  const float* W_hh1     = (const float*)d_in[6]; // [32,512,128]
  const float* b1        = (const float*)d_in[7]; // [32,512]
  const float* W_out     = (const float*)d_in[8]; // [32,128]
  const float* b_out     = (const float*)d_in[9]; // [32]
  float* out = (float*)d_out;                     // [4096,32]

  // workspace: G fp32 [D,T,512] (256MB, reused for both layers) + h1 f16 [T,D,128]
  float*    G  = (float*)d_ws;
  _Float16* h1 = (_Float16*)((char*)d_ws + (size_t)NDEV * TSEQ * NG * sizeof(float));

  const dim3 gridG(TSEQ / 128, NG / 128, NDEV);
  const size_t ldsA = (size_t)2 * 128 * (96 + 8) * 2;    // 53,248 B
  const size_t ldsC = (size_t)2 * 128 * (128 + 8) * 2;   // 69,632 B
  const size_t ldsR = (size_t)512 * 136 * 2 + 2048 + 512 + 320; // 142,144 B

  inproj_gemm<72, 3, true ><<<gridG, 256, ldsA, stream>>>(shared_in, per_dev, nullptr, W_ih0, b0, G);
  lstm_recur<false><<<NDEV, 256, ldsR, stream>>>(G, W_hh0, h1, nullptr, nullptr, nullptr);
  inproj_gemm<128, 4, false><<<gridG, 256, ldsC, stream>>>(nullptr, nullptr, h1, W_ih1, b1, G);
  lstm_recur<true ><<<NDEV, 256, ldsR, stream>>>(G, W_hh1, nullptr, W_out, b_out, out);
}
